// GruLayer_70231305224936
// MI455X (gfx1250) — matrix operand order repaired and checked by reading, not compiled, not run on hardware
//
#include <hip/hip_runtime.h>
#include <hip/hip_bf16.h>
#include <math.h>

typedef __attribute__((ext_vector_type(16))) _Float16 v16h;
typedef __attribute__((ext_vector_type(8)))  float    v8f;

// Problem dims (compile-time)
static constexpr int kB = 64;     // batch
static constexpr int kT = 512;    // time
static constexpr int kD = 768;    // input features
static constexpr int kH = 768;    // hidden
static constexpr int kG = 3 * kH; // gate width = 2304
static constexpr int kNT32 = kG / 32;          // 72 N super-tiles (32 wide)
static constexpr int kMT32_X = (kB * kT) / 32; // 1024 M super-tiles for input GEMM
static constexpr int kMT16_R = kB / 16;        // 4 M tiles (16 rows) for recurrent GEMM

// ---------------------------------------------------------------------------
// Register-blocked wave GEMM: RM x RN accumulators of 16x16 f32, fed by
// v_wmma_f32_16x16x32_f16.  A fragments loaded straight from row-major f32
// memory per the ISA 16-bit A layout (lane = M row, half-wave picks K-octet).
// B tile (32 x RN*16) staged through LDS with coalesced loads (row padded to
// RN*16+1 floats to keep half-wave fragment reads on disjoint banks), then
// read back per the ISA 16-bit B layout (lane = N col, half-wave picks K-octet).
// ---------------------------------------------------------------------------
template <int RM, int RN>
__device__ inline void wave_gemm_block(const float* __restrict__ A, int ldA,
                                       const float* __restrict__ Bm, int ldB,
                                       int m0, int n0, int K,
                                       float (* __restrict__ lb)[RN * 16 + 1],
                                       int lane, v8f acc[RM][RN])
{
    constexpr int W = RN * 16;
    const int mrow = lane & 15;
    const int koff = (lane >> 4) * 8;
    const int kb16 = (lane >> 4) * 16;
    const int bcol = lane & 15;

    for (int kk = 0; kk < K; kk += 32) {
        // ---- A fragments: contiguous 32B + 32B per lane per fragment ----
        v16h a[RM];
#pragma unroll
        for (int mi = 0; mi < RM; ++mi) {
            const float* ap =
                A + (size_t)(m0 + mi * 16 + mrow) * ldA + kk + koff;
            if (kk + 32 < K) __builtin_prefetch(ap + 32, 0, 3); // next K chunk
#pragma unroll
            for (int i = 0; i < 8; ++i) {
                a[mi][i]     = (_Float16)ap[i];
                a[mi][8 + i] = (_Float16)ap[16 + i];
            }
        }

        // ---- stage B 32 x W tile into LDS (coalesced 128B rows) ----
#pragma unroll
        for (int r = 0; r < W; ++r) {
            int e    = r * 32 + lane;   // 0 .. 32*W-1
            int brow = e / W;           // K within tile
            int bc   = e % W;           // N within tile
            lb[brow][bc] = Bm[(size_t)(kk + brow) * ldB + n0 + bc];
        }
        // cross-lane RAW through LDS within this wave: wait on DS counter
        asm volatile("s_wait_dscnt 0" ::: "memory");

        // ---- B fragments from LDS ----
        v16h b[RN];
#pragma unroll
        for (int ni = 0; ni < RN; ++ni)
#pragma unroll
            for (int j = 0; j < 16; ++j)
                b[ni][j] = (_Float16)lb[(j < 8 ? koff + j : 16 + koff + (j - 8))][ni * 16 + bcol];

        // ---- RM x RN WMMA outer product ----
#pragma unroll
        for (int mi = 0; mi < RM; ++mi)
#pragma unroll
            for (int ni = 0; ni < RN; ++ni)
                acc[mi][ni] = __builtin_amdgcn_wmma_f32_16x16x32_f16(
                    false, a[mi], false, b[ni], (short)0, acc[mi][ni],
                    false, false);
    }
}

// ---------------------------------------------------------------------------
// Phase 1: xp = x @ W + b[0]  (both directions).  One wave per 32x32 tile.
// grid.x = (1024*72)/4 blocks of 128 threads (4 waves), grid.y = direction.
// ---------------------------------------------------------------------------
__global__ void __launch_bounds__(128)
gru_xp_gemm(const float* __restrict__ x,
            const float* __restrict__ Wf, const float* __restrict__ bf,
            const float* __restrict__ Wb, const float* __restrict__ bb,
            float* __restrict__ xp_f, float* __restrict__ xp_b)
{
    __shared__ float lb[4][32][33];
    const int lane = threadIdx.x & 31;
    const int wave = threadIdx.x >> 5;
    const int dir  = blockIdx.y;

    const float* W    = dir ? Wb : Wf;
    const float* bias = dir ? bb : bf;       // row 0 of b
    float*       xp   = dir ? xp_b : xp_f;

    const int tile  = blockIdx.x * 4 + wave; // < 1024*72
    const int mTile = tile / kNT32;
    const int nTile = tile % kNT32;
    const int m0 = mTile * 32, n0 = nTile * 32;

    v8f acc[2][2] = {};
    wave_gemm_block<2, 2>(x, kD, W, kG, m0, n0, kD, lb[wave], lane, acc);

#pragma unroll
    for (int ni = 0; ni < 2; ++ni) {
        const int   n  = n0 + ni * 16 + (lane & 15);
        const float bv = bias[n];
#pragma unroll
        for (int mi = 0; mi < 2; ++mi) {
            const int mbase = m0 + mi * 16 + (lane >> 4) * 8;
#pragma unroll
            for (int r = 0; r < 8; ++r)
                xp[(size_t)(mbase + r) * kG + n] = acc[mi][ni][r] + bv;
        }
    }
}

// ---------------------------------------------------------------------------
// Phase 2a (per step): rec = h @ U + b[1]  (both directions).
// 16x32 wave tiles: 2 dirs * 4 * 72 = 576 waves -> 144 blocks of 4 waves.
// ---------------------------------------------------------------------------
__global__ void __launch_bounds__(128)
gru_rec_gemm(const float* __restrict__ h_f, const float* __restrict__ h_b,
             const float* __restrict__ Uf, const float* __restrict__ bf,
             const float* __restrict__ Ub, const float* __restrict__ bb,
             float* __restrict__ rec_f, float* __restrict__ rec_b)
{
    __shared__ float lb[4][32][33];
    const int lane = threadIdx.x & 31;
    const int wave = threadIdx.x >> 5;

    const int tile = blockIdx.x * 4 + wave;        // < 576
    const int dir  = tile / (kMT16_R * kNT32);     // 0 fwd, 1 bwd
    const int tt   = tile % (kMT16_R * kNT32);
    const int mTile = tt / kNT32;
    const int nTile = tt % kNT32;
    const int m0 = mTile * 16, n0 = nTile * 32;

    const float* h    = dir ? h_b : h_f;
    const float* U    = dir ? Ub : Uf;
    const float* bias = (dir ? bb : bf) + kG;      // row 1 of b (recurrent)
    float*       rec  = dir ? rec_b : rec_f;

    v8f acc[1][2] = {};
    wave_gemm_block<1, 2>(h, kH, U, kG, m0, n0, kH, lb[wave], lane, acc);

#pragma unroll
    for (int ni = 0; ni < 2; ++ni) {
        const int   n  = n0 + ni * 16 + (lane & 15);
        const float bv = bias[n];
        const int mbase = m0 + (lane >> 4) * 8;
#pragma unroll
        for (int r = 0; r < 8; ++r)
            rec[(size_t)(mbase + r) * kG + n] = acc[0][ni][r] + bv;
    }
}

// ---------------------------------------------------------------------------
// Phase 2b (per step): gates + state update + output slice.
// bwd direction uses t = T-1-s (equivalent to reference's reverse/re-reverse).
// ---------------------------------------------------------------------------
__global__ void __launch_bounds__(256)
gru_gates(const float* __restrict__ xp_f, const float* __restrict__ xp_b,
          const float* __restrict__ rec_f, const float* __restrict__ rec_b,
          float* __restrict__ h_f, float* __restrict__ h_b,
          float* __restrict__ out, int s)
{
    int idx = blockIdx.x * blockDim.x + threadIdx.x;
    if (idx >= 2 * kB * kH) return;
    const int dir = idx >= kB * kH;
    const int i   = idx - dir * kB * kH;
    const int b   = i / kH;
    const int n   = i % kH;
    const int t   = dir ? (kT - 1 - s) : s;

    const float* xp  = dir ? xp_b : xp_f;
    const float* rec = dir ? rec_b : rec_f;
    float*       h   = dir ? h_b : h_f;

    const size_t xoff = ((size_t)b * kT + t) * kG;
    const float xz = xp[xoff + n];
    const float xr = xp[xoff + kH + n];
    const float xh = xp[xoff + 2 * kH + n];

    const size_t roff = (size_t)b * kG;
    const float rz = rec[roff + n];
    const float rr = rec[roff + kH + n];
    const float rh = rec[roff + 2 * kH + n];

    const float z  = 1.0f / (1.0f + __expf(-(xz + rz)));
    const float r  = 1.0f / (1.0f + __expf(-(xr + rr)));
    const float hh = tanhf(xh + r * rh);
    const float hp = h[i];
    const float hn = z * hp + (1.0f - z) * hh;

    h[i] = hn;
    out[((size_t)b * kT + t) * (2 * kH) + dir * kH + n] = hn;
}

__global__ void gru_zero(float* __restrict__ p, int n)
{
    int i = blockIdx.x * blockDim.x + threadIdx.x;
    if (i < n) p[i] = 0.0f;
}

// ---------------------------------------------------------------------------
extern "C" void kernel_launch(void* const* d_in, const int* in_sizes, int n_in,
                              void* d_out, int out_size, void* d_ws, size_t ws_size,
                              hipStream_t stream)
{
    const float* x     = (const float*)d_in[0];
    const float* W_fwd = (const float*)d_in[1];
    const float* U_fwd = (const float*)d_in[2];
    const float* b_fwd = (const float*)d_in[3];
    const float* W_bwd = (const float*)d_in[4];
    const float* U_bwd = (const float*)d_in[5];
    const float* b_bwd = (const float*)d_in[6];
    float* out = (float*)d_out;

    // Workspace layout
    float* xp_f  = (float*)d_ws;
    float* xp_b  = xp_f  + (size_t)kB * kT * kG;
    float* rec_f = xp_b  + (size_t)kB * kT * kG;
    float* rec_b = rec_f + (size_t)kB * kG;
    float* h_f   = rec_b + (size_t)kB * kG;
    float* h_b   = h_f   + (size_t)kB * kH;

    // Zero hidden state (both directions)
    {
        int n = 2 * kB * kH;
        gru_zero<<<(n + 255) / 256, 256, 0, stream>>>(h_f, n);
    }

    // Phase 1: input projections, one wave per 32x32 tile.
    {
        dim3 grid((kMT32_X * kNT32) / 4, 2);
        gru_xp_gemm<<<grid, 128, 0, stream>>>(x, W_fwd, b_fwd, W_bwd, b_bwd,
                                              xp_f, xp_b);
    }

    // Phase 2: sequential recurrence, fwd + bwd fused per launch.
    for (int s = 0; s < kT; ++s) {
        gru_rec_gemm<<<(2 * kMT16_R * kNT32) / 4, 128, 0, stream>>>(
            h_f, h_b, U_fwd, b_fwd, U_bwd, b_bwd, rec_f, rec_b);
        gru_gates<<<(2 * kB * kH + 255) / 256, 256, 0, stream>>>(
            xp_f, xp_b, rec_f, rec_b, h_f, h_b, out, s);
    }
}
